// FullAttention_39281770889979
// MI455X (gfx1250) — matrix operand order repaired and checked by reading, not compiled, not run on hardware
//
#include <hip/hip_runtime.h>
#include <hip/hip_bf16.h>
#include <stdint.h>

typedef __bf16 bf16;
typedef __attribute__((ext_vector_type(16))) __bf16 v16bf;
typedef __attribute__((ext_vector_type(8)))  __bf16 v8bf;
typedef __attribute__((ext_vector_type(8)))  float  v8f;
typedef __attribute__((ext_vector_type(4)))  unsigned v4u;
typedef __attribute__((ext_vector_type(8)))  int      v8i;
typedef __attribute__((ext_vector_type(4)))  int      v4i;

union V16 { v16bf v; v8bf h[2]; };
union V8S { v8bf v; bf16 e[8]; };

constexpr int BATCH = 8, T = 2048, E = 2048, D = 256;

// ---------------------------------------------------------------------------
// TDM helper: 2D tile load (strided global rows -> packed LDS). Units are
// data_size elements; data_size_log: 0=1B,1=2B,2=4B,3=8B. tile_d1==0 -> 1D.
// D# per CDNA5 ISA 8.3/8.4; 6-arg builtin form (groups 2/3/4 zero, <=2D).
// ---------------------------------------------------------------------------
__device__ __forceinline__ void tdm_load_to_lds(
    unsigned lds_addr, const void* gptr, unsigned data_size_log,
    unsigned tile_d0, unsigned tile_d1,
    unsigned long long tensor_d0, unsigned long long tensor_d1,
    unsigned long long stride0)
{
    unsigned long long ga = (unsigned long long)(uintptr_t)gptr;
    v4u g0;
    g0[0] = 1u;                                                // count=1, user mode
    g0[1] = lds_addr;                                          // [63:32] lds byte addr
    g0[2] = (unsigned)(ga & 0xFFFFFFFFu);                      // global addr lo
    g0[3] = (unsigned)((ga >> 32) & 0x01FFFFFFu) | (2u << 30); // addr hi | type=2
    v8i g1;
    g1[0] = (int)(data_size_log << 16);                        // wg_mask=0, data_size
    g1[1] = (int)((tensor_d0 & 0xFFFFu) << 16);
    g1[2] = (int)(((tensor_d0 >> 16) & 0xFFFFu) | ((tensor_d1 & 0xFFFFu) << 16));
    g1[3] = (int)(((tensor_d1 >> 16) & 0xFFFFu) | ((tile_d0 & 0xFFFFu) << 16));
    g1[4] = (int)(tile_d1 & 0xFFFFu);                          // tile_dim1; dim2=0
    g1[5] = (int)(stride0 & 0xFFFFFFFFu);
    g1[6] = (int)((stride0 >> 32) & 0xFFFFu);
    g1[7] = 0;
    v4i z4 = {0, 0, 0, 0};
    v8i z8 = {0, 0, 0, 0, 0, 0, 0, 0};
    __builtin_amdgcn_tensor_load_to_lds(g0, g1, z4, z4, z8, 0);
}

__device__ __forceinline__ void tdm_wait() {
    __builtin_amdgcn_s_wait_tensorcnt(0);
}

// ---------------------------------------------------------------------------
// Prep A: x f32 -> bf16 [B*T][E], once per launch.
// ---------------------------------------------------------------------------
__global__ __launch_bounds__(256) void convert_x_kernel(
    const float* __restrict__ x, bf16* __restrict__ xb)
{
    size_t i = ((size_t)blockIdx.x * 256 + threadIdx.x) * 4;
    float4 f = *(const float4*)(x + i);
    bf16* d = xb + i;
    d[0] = (bf16)f.x; d[1] = (bf16)f.y; d[2] = (bf16)f.z; d[3] = (bf16)f.w;
}

// ---------------------------------------------------------------------------
// Prep B: W{q,k,v} f32 [E][D] -> transposed bf16 wt[3*D][E].
// ---------------------------------------------------------------------------
__global__ __launch_bounds__(256) void convert_w_kernel(
    const float* __restrict__ Wq, const float* __restrict__ Wk,
    const float* __restrict__ Wv, bf16* __restrict__ wt)
{
    const int EC = E / 4;
    int t = blockIdx.x * 256 + threadIdx.x;       // 3*D*E/4 threads
    int e = (t % EC) * 4;
    int rest = t / EC;
    int col = rest & (D - 1);
    int m   = rest >> 8;
    const float* W = (m == 0) ? Wq : (m == 1) ? Wk : Wv;
    bf16* dst = wt + ((size_t)m * D + col) * E + e;
#pragma unroll
    for (int j = 0; j < 4; ++j)
        dst[j] = (bf16)W[(size_t)(e + j) * D + col];
}

// ---------------------------------------------------------------------------
// Kernel 1: Q/K/V = x @ W via WMMA, double-buffered TDM staging (depth-64
// K-tiles; 48 KB LDS). grid (B*T/64, 6); 8 waves = 4 row-strips x 2 halves.
// V is stored transposed [B][D][T] with packed b128 stores.
// ---------------------------------------------------------------------------
__global__ __launch_bounds__(256) void qkv_proj_kernel(
    const bf16* __restrict__ xb, const bf16* __restrict__ wt,
    bf16* __restrict__ qws, bf16* __restrict__ kws, bf16* __restrict__ vtws)
{
    __shared__ bf16 Xl[2][64 * 64];    // 2 x 8 KB:  [row][k]
    __shared__ bf16 Wl[2][128 * 64];   // 2 x 16 KB: [col][k]

    const int tid   = threadIdx.x;
    const int lane  = tid & 31;
    const int wave  = tid >> 5;
    const int strip = wave >> 1;
    const int half  = wave & 1;
    const int hi    = lane >> 4;
    const int ln    = lane & 15;
    const int m0    = blockIdx.x * 64;
    const int cb    = blockIdx.y;                // 0..5 -> 128-col slab of [768]
    const int gcol0 = cb * 128;

    v8f acc[4] = {};

    const bf16* xsrc = xb + (size_t)m0 * E;
    const bf16* wsrc = wt + (size_t)gcol0 * E;

    // prologue: stage tile 0 into buffer 0
    if (wave == 0) {
        tdm_load_to_lds((unsigned)(uintptr_t)(&Xl[0][0]), xsrc, 1, 64, 64,
                        (unsigned long long)E, 64ull, (unsigned long long)E);
        tdm_load_to_lds((unsigned)(uintptr_t)(&Wl[0][0]), wsrc, 1, 64, 128,
                        (unsigned long long)E, 128ull, (unsigned long long)E);
        tdm_wait();
    }
    __syncthreads();

    const int NIT = E / 64;
    for (int i = 0; i < NIT; ++i) {
        const int p = i & 1;
        // prefetch tile i+1 into the other buffer (overlaps with compute)
        if (wave == 0 && i + 1 < NIT) {
            int e1 = (i + 1) * 64;
            tdm_load_to_lds((unsigned)(uintptr_t)(&Xl[p ^ 1][0]), xsrc + e1, 1,
                            64, 64, (unsigned long long)E, 64ull, (unsigned long long)E);
            tdm_load_to_lds((unsigned)(uintptr_t)(&Wl[p ^ 1][0]), wsrc + e1, 1,
                            64, 128, (unsigned long long)E, 128ull, (unsigned long long)E);
        }

#pragma unroll
        for (int ks = 0; ks < 64; ks += 32) {
            V16 a;
            const bf16* arow = &Xl[p][(strip * 16 + ln) * 64];
            a.h[0] = *(const v8bf*)(arow + ks + hi * 16);
            a.h[1] = *(const v8bf*)(arow + ks + hi * 16 + 8);
#pragma unroll
            for (int t4 = 0; t4 < 4; ++t4) {
                int col = half * 64 + t4 * 16 + ln;
                V16 bfr;
                bfr.v = *(const v16bf*)(&Wl[p][col * 64 + ks + hi * 16]);
                acc[t4] = __builtin_amdgcn_wmma_f32_16x16x32_bf16(
                    false, a.v, false, bfr.v, (short)0, acc[t4], false, false);
            }
        }
        if (wave == 0) tdm_wait();   // next tile landed (overlapped with wmma)
        __syncthreads();
    }

    // epilogue. C layout: VGPR r -> row r + 8*hi, col = ln.
    const int isV = (cb >= 4);
#pragma unroll
    for (int t4 = 0; t4 < 4; ++t4) {
        int col = (gcol0 & 255) + half * 64 + t4 * 16 + ln;   // col within D
        if (isV) {
            // rows r=0..7 are consecutive t at fixed d -> one b128 store
            int rowg0 = m0 + strip * 16 + hi * 8;
            int b = rowg0 >> 11, t0 = rowg0 & (T - 1);
            V8S pk;
#pragma unroll
            for (int r = 0; r < 8; ++r) pk.e[r] = (bf16)acc[t4][r];
            *(v8bf*)(vtws + ((size_t)b * D + col) * T + t0) = pk.v;
        } else {
            bf16* outp = (cb < 2) ? qws : kws;
#pragma unroll
            for (int r = 0; r < 8; ++r) {
                int rowg = m0 + strip * 16 + r + hi * 8;
                outp[(size_t)rowg * D + col] = (bf16)acc[t4][r];
            }
        }
    }
}

// ---------------------------------------------------------------------------
// Kernel 2: causal flash attention, double-buffered TDM (64 KB LDS exactly).
// grid (T/64, B); 8 waves = 4 strips x 2 column halves. P is staged in the
// dead half of the current K buffer. Net logit scale = 1/D.
// ---------------------------------------------------------------------------
__global__ __launch_bounds__(256) void flash_attn_kernel(
    const bf16* __restrict__ q, const bf16* __restrict__ k,
    const bf16* __restrict__ vt, float* __restrict__ out)
{
    __shared__ bf16 Kb[2][32 * 256];    // 2 x 16 KB keys [key][d]
    __shared__ bf16 Vb[2][256 * 32];    // 2 x 16 KB values [d][key]

    const int tid   = threadIdx.x;
    const int lane  = tid & 31;
    const int wave  = tid >> 5;
    const int strip = wave >> 1;
    const int half  = wave & 1;
    const int hi    = lane >> 4;
    const int ln    = lane & 15;
    const int q0    = blockIdx.x * 64;
    const int b     = blockIdx.y;
    const size_t base = (size_t)b * T * D;

    const bf16* ksrc = k + base;
    const bf16* vsrc = vt + (size_t)b * D * T;

    // Q fragments for this strip (8 x 16x32 A-frags in registers)
    V16 qf[8];
    {
        const bf16* qrow = q + base + (size_t)(q0 + strip * 16 + ln) * D;
#pragma unroll
        for (int f = 0; f < 8; ++f) {
            qf[f].h[0] = *(const v8bf*)(qrow + f * 32 + hi * 16);
            qf[f].h[1] = *(const v8bf*)(qrow + f * 32 + hi * 16 + 8);
        }
    }

    v8f oacc[8] = {};
    float mrow[8], lrow[8];
#pragma unroll
    for (int r = 0; r < 8; ++r) { mrow[r] = -3.0e38f; lrow[r] = 0.0f; }

    const float scale = 1.0f / 256.0f;
    const int NIT = (q0 >> 5) + 2;        // kv tiles of 32 keys, causal bound

    // prologue: stage kv-tile 0 into buffer 0
    if (wave == 0) {
        tdm_load_to_lds((unsigned)(uintptr_t)(&Kb[0][0]), ksrc, 3, 2048, 0,
                        2048ull, 1ull, 2048ull);
        tdm_load_to_lds((unsigned)(uintptr_t)(&Vb[0][0]), vsrc, 1, 32, 256,
                        (unsigned long long)T, 256ull, (unsigned long long)T);
        tdm_wait();
    }
    __syncthreads();

    for (int i = 0; i < NIT; ++i) {
        const int p = i & 1;
        const int kv0 = i * 32;

        // --- S = Q K^T: 2 key-subtiles x 8 ksteps ---
        v8f sacc[2] = {};
#pragma unroll
        for (int f = 0; f < 8; ++f) {
#pragma unroll
            for (int n = 0; n < 2; ++n) {
                V16 bfr;
                bfr.v = *(const v16bf*)(&Kb[p][(n * 16 + ln) * 256 + f * 32 + hi * 16]);
                sacc[n] = __builtin_amdgcn_wmma_f32_16x16x32_bf16(
                    false, qf[f].v, false, bfr.v, (short)0, sacc[n], false, false);
            }
        }

        // --- online softmax ---
        const bool needmask = (kv0 + 31 > q0);
        float pvv[2][8], newm[8];
#pragma unroll
        for (int r = 0; r < 8; ++r) newm[r] = mrow[r];
#pragma unroll
        for (int n = 0; n < 2; ++n)
#pragma unroll
            for (int r = 0; r < 8; ++r) {
                float s = sacc[n][r] * scale;
                int colg = kv0 + n * 16 + ln;
                int rowg = q0 + strip * 16 + r + hi * 8;
                if (needmask && colg > rowg) s = -3.0e38f;
                pvv[n][r] = s;
                newm[r] = fmaxf(newm[r], s);
            }
#pragma unroll
        for (int r = 0; r < 8; ++r) {
            float m_ = newm[r];
            m_ = fmaxf(m_, __shfl_xor(m_, 1, 32));
            m_ = fmaxf(m_, __shfl_xor(m_, 2, 32));
            m_ = fmaxf(m_, __shfl_xor(m_, 4, 32));
            m_ = fmaxf(m_, __shfl_xor(m_, 8, 32));
            newm[r] = m_;
        }
        float rsum[8];
#pragma unroll
        for (int r = 0; r < 8; ++r) rsum[r] = 0.0f;
#pragma unroll
        for (int n = 0; n < 2; ++n)
#pragma unroll
            for (int r = 0; r < 8; ++r) {
                float pp = __expf(pvv[n][r] - newm[r]);
                pvv[n][r] = pp;
                rsum[r] += pp;
            }
#pragma unroll
        for (int r = 0; r < 8; ++r) {
            float s_ = rsum[r];
            s_ += __shfl_xor(s_, 1, 32);
            s_ += __shfl_xor(s_, 2, 32);
            s_ += __shfl_xor(s_, 4, 32);
            s_ += __shfl_xor(s_, 8, 32);
            rsum[r] = s_;
        }
#pragma unroll
        for (int r = 0; r < 8; ++r) {
            float alpha = __expf(mrow[r] - newm[r]);
            mrow[r] = newm[r];
            lrow[r] = lrow[r] * alpha + rsum[r];
#pragma unroll
            for (int t4 = 0; t4 < 8; ++t4) oacc[t4][r] *= alpha;
        }

        __syncthreads();   // (A) all waves done reading Kb[p]; it is now dead

        // P -> dead K buffer (C layout -> A layout via LDS)
        if (half == 0) {
            bf16* pdst = &Kb[p][strip * 16 * 32];
#pragma unroll
            for (int n = 0; n < 2; ++n)
#pragma unroll
                for (int r = 0; r < 8; ++r)
                    pdst[(r + hi * 8) * 32 + n * 16 + ln] = (bf16)pvv[n][r];
        }
        // prefetch next kv tile into the other buffer (overlaps with PV)
        if (wave == 0 && i + 1 < NIT) {
            int kv1 = (i + 1) * 32;
            tdm_load_to_lds((unsigned)(uintptr_t)(&Kb[p ^ 1][0]),
                            ksrc + (size_t)kv1 * D, 3, 2048, 0, 2048ull, 1ull, 2048ull);
            tdm_load_to_lds((unsigned)(uintptr_t)(&Vb[p ^ 1][0]),
                            vsrc + kv1, 1, 32, 256,
                            (unsigned long long)T, 256ull, (unsigned long long)T);
        }
        __syncthreads();   // (B) P visible

        // --- O += P @ V (K=32, one A frag; 8 x 16-col output tiles) ---
        V16 pa;
        {
            const bf16* prow = &Kb[p][strip * 16 * 32 + ln * 32];
            pa.h[0] = *(const v8bf*)(prow + hi * 16);
            pa.h[1] = *(const v8bf*)(prow + hi * 16 + 8);
        }
#pragma unroll
        for (int t4 = 0; t4 < 8; ++t4) {
            int d = half * 128 + t4 * 16 + ln;
            V16 bfr;
            bfr.v = *(const v16bf*)(&Vb[p][d * 32 + hi * 16]);
            oacc[t4] = __builtin_amdgcn_wmma_f32_16x16x32_bf16(
                false, pa.v, false, bfr.v, (short)0, oacc[t4], false, false);
        }

        if (wave == 0) tdm_wait();   // next tiles landed (overlapped with PV)
        __syncthreads();             // (C) safe to reuse buffers
    }

    // --- normalize and store f32 ---
#pragma unroll
    for (int r = 0; r < 8; ++r) {
        float inv = 1.0f / lrow[r];
        int rowg = q0 + strip * 16 + r + hi * 8;
        float* orow = out + base + (size_t)rowg * D;
#pragma unroll
        for (int t4 = 0; t4 < 8; ++t4)
            orow[half * 128 + t4 * 16 + ln] = oacc[t4][r] * inv;
    }
}

// ---------------------------------------------------------------------------
extern "C" void kernel_launch(void* const* d_in, const int* in_sizes, int n_in,
                              void* d_out, int out_size, void* d_ws, size_t ws_size,
                              hipStream_t stream) {
    const float* x  = (const float*)d_in[0];
    const float* Wq = (const float*)d_in[1];
    const float* Wk = (const float*)d_in[2];
    const float* Wv = (const float*)d_in[3];
    float* out = (float*)d_out;

    // workspace layout (bf16): xb[B*T*E] | wt[3*D*E] | q | k | vt
    bf16* xb  = (bf16*)d_ws;
    bf16* wtb = xb  + (size_t)BATCH * T * E;
    bf16* qws = wtb + (size_t)3 * D * E;
    bf16* kws = qws + (size_t)BATCH * T * D;
    bf16* vtw = kws + (size_t)BATCH * T * D;

    size_t nx = (size_t)BATCH * T * E;
    convert_x_kernel<<<(unsigned)(nx / 4 / 256), 256, 0, stream>>>(x, xb);
    convert_w_kernel<<<(unsigned)(3 * D * (E / 4) / 256), 256, 0, stream>>>(Wq, Wk, Wv, wtb);

    dim3 g1(BATCH * T / 64, 6);
    qkv_proj_kernel<<<g1, 256, 0, stream>>>(xb, wtb, qws, kws, vtw);

    dim3 g2(T / 64, BATCH);
    flash_attn_kernel<<<g2, 256, 0, stream>>>(qws, kws, vtw, out);
}
